// SRModel_84894323573213
// MI455X (gfx1250) — matrix operand order repaired and checked by reading, not compiled, not run on hardware
//
#include <hip/hip_runtime.h>
#include <hip/hip_bf16.h>

typedef _Float16 half_t;
typedef __attribute__((ext_vector_type(16))) _Float16 v16h;
typedef __attribute__((ext_vector_type(8)))  _Float16 v8h;
typedef __attribute__((ext_vector_type(8)))  float    v8f;
typedef __attribute__((ext_vector_type(4)))  int      v4i;

#define LRES 96
#define HRES 384
#define NLR  (LRES * LRES)     // 9216
#define NPIX (HRES * HRES)     // 147456
#define FEAT 64
#define HID  256
#define KK   49
#define NHEADS 4

#define WSTRIDE 264            // padded row stride (halves) for LDS tiles

#ifndef __has_builtin
#define __has_builtin(x) 0
#endif
#if __has_builtin(__builtin_amdgcn_global_load_async_to_lds_b128)
#define HAVE_ASYNC_LDS 1
#else
#define HAVE_ASYNC_LDS 0
#endif

#if HAVE_ASYNC_LDS
typedef __attribute__((address_space(1))) v4i g_v4i;   // global-AS int4
typedef __attribute__((address_space(3))) v4i l_v4i;   // LDS-AS int4
#endif

// Branchless GELU (tanh form), built on v_exp_f32 / v_rcp_f32.
__device__ __forceinline__ float gelu_f(float x) {
    const float x2 = x * x;
    const float u  = x * (0.7978845608f + 0.03567740814f * x2); // c*(x+0.044715x^3)
    const float e  = __builtin_amdgcn_exp2f(u * 2.885390082f);  // exp(2u)
    const float t  = 1.0f - 2.0f * __builtin_amdgcn_rcpf(e + 1.0f);
    return 0.5f * x * (1.0f + t);
}
__device__ __forceinline__ float fast_exp(float x) {
    return __builtin_amdgcn_exp2f(x * 1.4426950408889634f);
}
__device__ __forceinline__ int iclampi(int v, int lo, int hi) {
    return v < lo ? lo : (v > hi ? hi : v);
}

union AF { v16h v; v8h h[2]; };

// 16B global -> LDS stage: async path (ASYNCcnt) if available, sync fallback.
__device__ __forceinline__ void stage16(const void* g, void* l) {
#if HAVE_ASYNC_LDS
    __builtin_amdgcn_global_load_async_to_lds_b128(
        (g_v4i*)(unsigned long long)(uintptr_t)g,
        (l_v4i*)(unsigned int)(uintptr_t)l,
        0, 0);
#else
    *(v4i*)l = *(const v4i*)g;
#endif
}
__device__ __forceinline__ void wait_stage() {
#if HAVE_ASYNC_LDS
#if __has_builtin(__builtin_amdgcn_s_wait_asynccnt)
    __builtin_amdgcn_s_wait_asynccnt(0);
#else
    asm volatile("s_wait_asynccnt 0" ::: "memory");
#endif
#endif
}

// ---------------------------------------------------------------- encoder 3x3
__global__ __launch_bounds__(64) void k_conv3x3(
    const float* __restrict__ in, const float* __restrict__ wgt,
    const float* __restrict__ bias, float* __restrict__ out, int IC)
{
    __shared__ float patch[FEAT * 9];
    const int pix = blockIdx.x;
    const int y = pix / LRES, x = pix % LRES;
    const int tid = threadIdx.x;
    for (int i = tid; i < IC * 9; i += 64) {
        const int ic = i / 9, k = i % 9;
        const int yy = y + k / 3 - 1, xx = x + k % 3 - 1;
        patch[i] = (yy >= 0 && yy < LRES && xx >= 0 && xx < LRES)
                       ? in[ic * NLR + yy * LRES + xx] : 0.0f;
    }
    __syncthreads();
    float a = bias[tid];
    for (int ic = 0; ic < IC; ++ic) {
        const float* wp = wgt + (tid * IC + ic) * 9;
        const float* pp = patch + ic * 9;
#pragma unroll
        for (int k = 0; k < 9; ++k) a = fmaf(wp[k], pp[k], a);
    }
    out[tid * NLR + pix] = gelu_f(a);
}

// ---------------------------------------------------------------- cond 1x1
__global__ __launch_bounds__(64) void k_conv1x1(
    const float* __restrict__ in, const float* __restrict__ wgt,
    const float* __restrict__ bias, float* __restrict__ out,
    int ic_stride, int gelu_in)
{
    __shared__ float col[FEAT];
    const int pix = blockIdx.x;
    const int tid = threadIdx.x;
    const float v = in[tid * NLR + pix];
    col[tid] = gelu_in ? gelu_f(v) : v;
    __syncthreads();
    float a = bias[tid];
#pragma unroll 4
    for (int ic = 0; ic < FEAT; ++ic)
        a = fmaf(wgt[tid * ic_stride + ic], col[ic], a);
    out[tid * NLR + pix] = a;
}

// ------------------------------------------- MLP weight prep: f32 -> f16, W^T
__global__ void k_prep_wt(
    const float* __restrict__ w0, const float* __restrict__ w1,
    const float* __restrict__ w2, const float* __restrict__ w3,
    const float* __restrict__ w4, const float* __restrict__ hk,
    const float* __restrict__ hg, half_t* __restrict__ wt)
{
    const int i = blockIdx.x * blockDim.x + threadIdx.x;
    if (i >= 6 * 65536) return;
    const int L = i >> 16, e = i & 65535;
    const int n = e >> 8, k = e & 255;
    float v = 0.0f;
    if (L == 0)      { if (k < 106) v = w0[k * HID + n]; }
    else if (L == 1) v = w1[k * HID + n];
    else if (L == 2) v = w2[k * HID + n];
    else if (L == 3) v = w3[k * HID + n];
    else if (L == 4) v = w4[k * HID + n];
    else {
        if (n < 196)      v = hk[k * 196 + n];
        else if (n < 200) v = hg[k * 4 + (n - 196)];
    }
    wt[i] = (half_t)v;
}

__global__ void k_prep_bh(const float* __restrict__ hkb,
                          const float* __restrict__ hgb, float* __restrict__ bh)
{
    const int n = threadIdx.x;  // 256
    bh[n] = (n < 196) ? hkb[n] : (n < 200 ? hgb[n - 196] : 0.0f);
}

// ---------------------------------------------------------------- bilinear up
__global__ void k_upsample(const float* __restrict__ lr, float* __restrict__ up)
{
    const int i = blockIdx.x * blockDim.x + threadIdx.x;
    if (i >= 3 * NPIX) return;
    const int c = i / NPIX, p = i - c * NPIX;
    const int y = p / HRES, x = p % HRES;
    const float sx = (x + 0.5f) * 0.25f - 0.5f;
    const float sy = (y + 0.5f) * 0.25f - 0.5f;
    const float fxf = floorf(sx), fyf = floorf(sy);
    const float fx = sx - fxf, fy = sy - fyf;
    const int x0 = (int)fxf, y0 = (int)fyf;
    const int x0c = iclampi(x0, 0, LRES - 1), x1c = iclampi(x0 + 1, 0, LRES - 1);
    const int y0c = iclampi(y0, 0, LRES - 1), y1c = iclampi(y0 + 1, 0, LRES - 1);
    const float* pl = lr + c * NLR;
    const float v00 = pl[y0c * LRES + x0c], v01 = pl[y0c * LRES + x1c];
    const float v10 = pl[y1c * LRES + x0c], v11 = pl[y1c * LRES + x1c];
    up[i] = (1.0f - fy) * ((1.0f - fx) * v00 + fx * v01)
          +         fy  * ((1.0f - fx) * v10 + fx * v11);
}

// ------------------------------------------------ fused per-pixel MLP (WMMA)
// 256 threads = 8 waves; each wave carries 32 pixels (two 16-row A tiles) so
// every B fragment read from LDS feeds two WMMAs (halves LDS read traffic).
// Dynamic LDS pool: [0,135168)  weight tile  [256][WSTRIDE] f16 (async-staged)
//                   [135168,..) per-wave act [32][WSTRIDE]  f16
__global__ __launch_bounds__(256) void k_mlp_wmma(
    const float* __restrict__ feat, const float* __restrict__ lr,
    const half_t* __restrict__ wt,
    const float* __restrict__ b0, const float* __restrict__ b1,
    const float* __restrict__ b2, const float* __restrict__ b3,
    const float* __restrict__ b4, const float* __restrict__ bh,
    float* __restrict__ sr)
{
    extern __shared__ char smem[];
    half_t* wbuf = (half_t*)smem;                                   // [256][264]
    const int tid = threadIdx.x;
    const int wv = tid >> 5, lane = tid & 31;
    half_t (*act16)[WSTRIDE] =
        (half_t (*)[WSTRIDE])(smem + 135168 + wv * (32 * WSTRIDE * 2));
    const int p0 = (blockIdx.x * 8 + wv) * 32;

    // ---- build z features [32 px][256] in f16 (cols 106..255 = 0) ----
    for (int i = lane; i < 32 * 256; i += 32) {
        const int r = i >> 8, c = i & 255;
        const int p = p0 + r;
        const int x = p % HRES, y = p / HRES;
        const float lx = (x + 0.5f) * 0.25f - 0.5f;
        const float ly = (y + 0.5f) * 0.25f - 0.5f;
        const float cxf = fminf(fmaxf(rintf(lx), 0.0f), (float)(LRES - 1));
        const float cyf = fminf(fmaxf(rintf(ly), 0.0f), (float)(LRES - 1));
        float v = 0.0f;
        if (c < FEAT) {
            v = feat[c * NLR + (int)cyf * LRES + (int)cxf];
        } else if (c < 104) {
            const int j = c - 64, b = j >> 2, t = j & 3;
            const float fr = exp2f((float)b) * 3.14159265358979323846f;
            const float xn = (x + 0.5f) * (2.0f / HRES) - 1.0f;
            const float yn = (y + 0.5f) * (2.0f / HRES) - 1.0f;
            const float arg = ((t < 2) ? xn : yn) * fr;
            v = ((t & 1) == 0) ? sinf(arg) : cosf(arg);
        } else if (c == 104) {
            v = lx - cxf;
        } else if (c == 105) {
            v = ly - cyf;
        }
        act16[r][c] = (half_t)v;
    }
    __syncthreads();

    const float* const biases[6] = { b0, b1, b2, b3, b4, bh };
    const int row = lane & 15, seg = lane >> 4;

#pragma unroll
    for (int L = 0; L < 6; ++L) {
        // ---- stage this layer's 128KB weight tile into LDS (async) ----
        {
            const char* gsrc = (const char*)(wt + (size_t)L * 65536);
            char* ldst = (char*)wbuf;
#pragma unroll
            for (int c2 = 0; c2 < 32; ++c2) {
                const int c = tid + c2 * 256;          // 8192 16B chunks
                const int rowi = c >> 5, o = c & 31;
                stage16(gsrc + rowi * 512 + o * 16,
                        ldst + rowi * (WSTRIDE * 2) + o * 16);
            }
        }
        // ---- A fragments (16-bit A 16x32 layout), two M-tiles in regs ----
        AF afr0[8], afr1[8];
#pragma unroll
        for (int kt = 0; kt < 8; ++kt) {
            const half_t* ap0 = &act16[row][kt * 32 + seg * 8];
            const half_t* ap1 = &act16[16 + row][kt * 32 + seg * 8];
            afr0[kt].h[0] = *(const v8h*)ap0;
            afr0[kt].h[1] = *(const v8h*)(ap0 + 16);
            afr1[kt].h[0] = *(const v8h*)ap1;
            afr1[kt].h[1] = *(const v8h*)(ap1 + 16);
        }
        wait_stage();
        __syncthreads();    // weight tile visible to all waves

        const float* __restrict__ bs = biases[L];
        for (int nt = 0; nt < 16; ++nt) {
            const int ncol = nt * 16 + row;
            const float bv = bs[ncol];
            v8f acc0 = { bv, bv, bv, bv, bv, bv, bv, bv };
            v8f acc1 = acc0;
            // B 32x16 layout from LDS tile: lane = N col, seg -> K+8..15 / K+24..31
            const half_t* wb = wbuf + ncol * WSTRIDE + seg * 8;
#pragma unroll
            for (int kt = 0; kt < 8; ++kt) {
                AF bf;
                bf.h[0] = *(const v8h*)(wb + kt * 32);
                bf.h[1] = *(const v8h*)(wb + kt * 32 + 16);
                acc0 = __builtin_amdgcn_wmma_f32_16x16x32_f16(
                    false, afr0[kt].v, false, bf.v, (short)0, acc0, false, false);
                acc1 = __builtin_amdgcn_wmma_f32_16x16x32_f16(
                    false, afr1[kt].v, false, bf.v, (short)0, acc1, false, false);
            }
            // C/D layout: vgpr g -> M = g + 8*seg (+16 for tile1), lane -> N
#pragma unroll
            for (int g = 0; g < 8; ++g) {
                float v0 = acc0[g], v1 = acc1[g];
                if (L < 5) { v0 = gelu_f(v0); v1 = gelu_f(v1); }
                act16[g + 8 * seg][ncol]      = (half_t)v0;
                act16[16 + g + 8 * seg][ncol] = (half_t)v1;
            }
        }
        __syncthreads();    // all waves done with wbuf + act stores complete
    }

    // ---- epilogue: head softmax, kernel mix, 7x7 gather -> sr (1 px/lane) ----
    float* kmix = (float*)smem + wv * (32 * KK);   // wbuf region is free now
    {
        const int r = lane, p = p0 + r;
        const int x = p % HRES, y = p / HRES;
        const float lx = (x + 0.5f) * 0.25f - 0.5f;
        const float ly = (y + 0.5f) * 0.25f - 0.5f;
        const int cx = iclampi((int)rintf(lx), 0, LRES - 1);
        const int cy = iclampi((int)rintf(ly), 0, LRES - 1);
        float gl[NHEADS];
        float gm = -1e30f;
#pragma unroll
        for (int h = 0; h < NHEADS; ++h) {
            gl[h] = (float)act16[r][196 + h];
            gm = fmaxf(gm, gl[h]);
        }
        float gs = 0.0f;
#pragma unroll
        for (int h = 0; h < NHEADS; ++h) { gl[h] = fast_exp(gl[h] - gm); gs += gl[h]; }
        for (int k = 0; k < KK; ++k) kmix[r * KK + k] = 0.0f;
        for (int h = 0; h < NHEADS; ++h) {
            float km = -1e30f;
            for (int k = 0; k < KK; ++k)
                km = fmaxf(km, (float)act16[r][h * KK + k]);
            float ks = 0.0f;
            for (int k = 0; k < KK; ++k)
                ks += fast_exp((float)act16[r][h * KK + k] - km);
            const float gw = gl[h] * __builtin_amdgcn_rcpf(gs * ks);
            for (int k = 0; k < KK; ++k)
                kmix[r * KK + k] += gw * fast_exp((float)act16[r][h * KK + k] - km);
        }
        float pr0 = 0.0f, pr1 = 0.0f, pr2 = 0.0f;
        for (int k = 0; k < KK; ++k) {
            const int nx = iclampi(cx + k % 7 - 3, 0, LRES - 1);
            const int ny = iclampi(cy + k / 7 - 3, 0, LRES - 1);
            const int gi = ny * LRES + nx;
            const float wk = kmix[r * KK + k];
            pr0 = fmaf(wk, lr[gi], pr0);
            pr1 = fmaf(wk, lr[NLR + gi], pr1);
            pr2 = fmaf(wk, lr[2 * NLR + gi], pr2);
        }
        sr[p] = pr0; sr[NPIX + p] = pr1; sr[2 * NPIX + p] = pr2;
    }
}

// --------------------------------- grouped-conv refiner + gray fusion (tiled)
__global__ __launch_bounds__(128) void k_refine_fuse(
    const float* __restrict__ up, const float* __restrict__ sr,
    const float* __restrict__ w0, const float* __restrict__ b0,
    const float* __restrict__ w1, const float* __restrict__ b1,
    const float* __restrict__ w2, const float* __restrict__ b2,
    float* __restrict__ out)
{
    __shared__ float lu[3][14][14];
    __shared__ float st[3][14][14];
    __shared__ float r1a[48][12][12];
    __shared__ float r1b[48][10][10];
    const int X0 = (blockIdx.x % 48) * 8, Y0 = (blockIdx.x / 48) * 8;
    const int tid = threadIdx.x;
    for (int i = tid; i < 3 * 196; i += 128) {
        const int c = i / 196, rr = i % 196;
        const int yy = rr / 14, xx = rr % 14;
        const int gy = Y0 - 3 + yy, gx = X0 - 3 + xx;
        const bool ok = (gx >= 0 && gx < HRES && gy >= 0 && gy < HRES);
        const int gi = c * NPIX + gy * HRES + gx;
        lu[c][yy][xx] = ok ? up[gi] : 0.0f;
        st[c][yy][xx] = ok ? sr[gi] : 0.0f;
    }
    __syncthreads();
    for (int i = tid; i < 48 * 144; i += 128) {        // layer0: 2->16 per group
        const int ch = i / 144, rr = i % 144;
        const int yy = rr / 12, xx = rr % 12;
        const int gy = Y0 - 2 + yy, gx = X0 - 2 + xx;
        float a = 0.0f;
        if (gx >= 0 && gx < HRES && gy >= 0 && gy < HRES) {
            const int g = ch >> 4;
            a = b0[ch];
#pragma unroll
            for (int ii = 0; ii < 2; ++ii) {
                const int gc = 2 * g + ii;  // x6 channel: [lu0..2, res0..2]
#pragma unroll
                for (int k = 0; k < 9; ++k) {
                    const int ay = yy + k / 3, ax = xx + k % 3;
                    const float xv = (gc < 3) ? lu[gc][ay][ax]
                                              : st[gc - 3][ay][ax] - lu[gc - 3][ay][ax];
                    a = fmaf(w0[(ch * 2 + ii) * 9 + k], xv, a);
                }
            }
            a = gelu_f(a);
        }
        r1a[ch][yy][xx] = a;
    }
    __syncthreads();
    for (int i = tid; i < 48 * 100; i += 128) {        // layer1: 16->16 per group
        const int ch = i / 100, rr = i % 100;
        const int yy = rr / 10, xx = rr % 10;
        const int gy = Y0 - 1 + yy, gx = X0 - 1 + xx;
        float a = 0.0f;
        if (gx >= 0 && gx < HRES && gy >= 0 && gy < HRES) {
            const int g = ch >> 4;
            a = b1[ch];
            for (int ii = 0; ii < 16; ++ii) {
                const float* wp = w1 + (ch * 16 + ii) * 9;
                const int sc = g * 16 + ii;
#pragma unroll
                for (int k = 0; k < 9; ++k)
                    a = fmaf(wp[k], r1a[sc][yy + k / 3][xx + k % 3], a);
            }
            a = gelu_f(a);
        }
        r1b[ch][yy][xx] = a;
    }
    __syncthreads();
    if (tid < 64) {                                    // layer2 + gray fusion
        const int yy = tid / 8, xx = tid % 8;
        const int gy = Y0 + yy, gx = X0 + xx;
        float res[3];
#pragma unroll
        for (int c = 0; c < 3; ++c) {
            float a = b2[c];
            for (int ii = 0; ii < 16; ++ii) {
                const float* wp = w2 + (c * 16 + ii) * 9;
                const int sc = c * 16 + ii;
#pragma unroll
                for (int k = 0; k < 9; ++k)
                    a = fmaf(wp[k], r1b[sc][yy + k / 3][xx + k % 3], a);
            }
            res[c] = a;
        }
        const int ay = yy + 3, ax = xx + 3;
        const float gcf[3] = { 0.2989f, 0.587f, 0.114f };
        float delta = 0.0f;
#pragma unroll
        for (int c = 0; c < 3; ++c)
            delta += gcf[c] * (st[c][ay][ax] - lu[c][ay][ax] + res[c]);
#pragma unroll
        for (int c = 0; c < 3; ++c) {
            float o = lu[c][ay][ax] + delta;
            o = fminf(fmaxf(o, 0.0f), 1.0f);
            out[c * NPIX + gy * HRES + gx] = o;
        }
    }
}

// ---------------------------------------------------------------------------
extern "C" void kernel_launch(void* const* d_in, const int* in_sizes, int n_in,
                              void* d_out, int out_size, void* d_ws, size_t ws_size,
                              hipStream_t stream)
{
    const float* lr      = (const float*)d_in[0];
    const float* enc_w0  = (const float*)d_in[1];
    const float* enc_b0  = (const float*)d_in[2];
    const float* enc_w1  = (const float*)d_in[3];
    const float* enc_b1  = (const float*)d_in[4];
    const float* enc_w2  = (const float*)d_in[5];
    const float* enc_b2  = (const float*)d_in[6];
    const float* enc_w3  = (const float*)d_in[7];
    const float* enc_b3  = (const float*)d_in[8];
    const float* cond_w0 = (const float*)d_in[9];
    const float* cond_b0 = (const float*)d_in[10];
    const float* cond_w1 = (const float*)d_in[11];
    const float* cond_b1 = (const float*)d_in[12];
    const float* mlp_w0  = (const float*)d_in[13];
    const float* mlp_b0  = (const float*)d_in[14];
    const float* mlp_w1  = (const float*)d_in[15];
    const float* mlp_b1  = (const float*)d_in[16];
    const float* mlp_w2  = (const float*)d_in[17];
    const float* mlp_b2  = (const float*)d_in[18];
    const float* mlp_w3  = (const float*)d_in[19];
    const float* mlp_b3  = (const float*)d_in[20];
    const float* mlp_w4  = (const float*)d_in[21];
    const float* mlp_b4  = (const float*)d_in[22];
    const float* hk_w    = (const float*)d_in[23];
    const float* hk_b    = (const float*)d_in[24];
    const float* hg_w    = (const float*)d_in[25];
    const float* hg_b    = (const float*)d_in[26];
    const float* ref_w0  = (const float*)d_in[27];
    const float* ref_b0  = (const float*)d_in[28];
    const float* ref_w1  = (const float*)d_in[29];
    const float* ref_b1  = (const float*)d_in[30];
    const float* ref_w2  = (const float*)d_in[31];
    const float* ref_b2  = (const float*)d_in[32];

    char* ws = (char*)d_ws;
    float*  featA = (float*)(ws);                       // 64*9216 f32
    float*  featB = (float*)(ws + 2359296);             // 64*9216 f32
    float*  lr_up = (float*)(ws + 4718592);             // 3*147456 f32
    float*  srb   = (float*)(ws + 6488064);             // 3*147456 f32
    half_t* wt    = (half_t*)(ws + 8257536);            // 6*256*256 f16
    float*  bh    = (float*)(ws + 9043968);             // 256 f32

    // local encoder
    k_conv3x3<<<NLR, 64, 0, stream>>>(lr,    enc_w0, enc_b0, featA, 3);
    k_conv3x3<<<NLR, 64, 0, stream>>>(featA, enc_w1, enc_b1, featB, 64);
    k_conv3x3<<<NLR, 64, 0, stream>>>(featB, enc_w2, enc_b2, featA, 64);
    k_conv3x3<<<NLR, 64, 0, stream>>>(featA, enc_w3, enc_b3, featB, 64);
    // cond fuser (cond channels are zero at inference -> use first 64 in-ch)
    k_conv1x1<<<NLR, 64, 0, stream>>>(featB, cond_w0, cond_b0, featA, 66, 0);
    k_conv1x1<<<NLR, 64, 0, stream>>>(featA, cond_w1, cond_b1, featB, 64, 1);
    // MLP weight prep + bilinear upsample
    k_prep_wt<<<(6 * 65536) / 256, 256, 0, stream>>>(mlp_w0, mlp_w1, mlp_w2,
                                                     mlp_w3, mlp_w4, hk_w, hg_w, wt);
    k_prep_bh<<<1, 256, 0, stream>>>(hk_b, hg_b, bh);
    k_upsample<<<(3 * NPIX + 255) / 256, 256, 0, stream>>>(lr, lr_up);
    // fused WMMA MLP + kernel prediction (270336B dynamic LDS pool)
    k_mlp_wmma<<<576, 256, 270336, stream>>>(featB, lr, wt, mlp_b0, mlp_b1,
                                             mlp_b2, mlp_b3, mlp_b4, bh, srb);
    // refiner + color-preserving fusion
    k_refine_fuse<<<48 * 48, 128, 0, stream>>>(lr_up, srb, ref_w0, ref_b0,
                                               ref_w1, ref_b1, ref_w2, ref_b2,
                                               (float*)d_out);
}
